// DeformableTransformerDecoderLayer_Det_81612968558767
// MI455X (gfx1250) — matrix operand order repaired and checked by reading, not compiled, not run on hardware
//
#include <hip/hip_runtime.h>
#include <cstddef>

// ---------------------------------------------------------------------------
// Types / helpers
// ---------------------------------------------------------------------------
typedef __attribute__((ext_vector_type(16))) __bf16 v16bf;
typedef __attribute__((ext_vector_type(8)))  float  v8f;

__device__ __forceinline__ unsigned short f2bf(float f) {
  unsigned u = __builtin_bit_cast(unsigned, f);
  u += 0x7FFFu + ((u >> 16) & 1u);           // round-to-nearest-even
  return (unsigned short)(u >> 16);
}
__device__ __forceinline__ float bf2f(unsigned short h) {
  unsigned u = ((unsigned)h) << 16;
  return __builtin_bit_cast(float, u);
}

// pack 4 floats -> 4 bf16, one aligned 8-byte LDS store
__device__ __forceinline__ void store_bf4(unsigned short* p, float4 v) {
  unsigned u0 = (unsigned)f2bf(v.x) | ((unsigned)f2bf(v.y) << 16);
  unsigned u1 = (unsigned)f2bf(v.z) | ((unsigned)f2bf(v.w) << 16);
  unsigned long long q = (unsigned long long)u0 | ((unsigned long long)u1 << 32);
  *(unsigned long long*)p = q;
}

union FragBF { unsigned u[8]; v16bf v; };

// A-fragment (16x32 bf16): lane m = lane&15; VGPR0-3 = K {half*8 .. half*8+7},
// VGPR4-7 = K {16+half*8 .. 16+half*8+7}  (ISA 7.12.2 16-bit A-matrix table)
__device__ __forceinline__ v16bf frag_a(const unsigned short* tile, int ld, int row0,
                                        int koff, int lane) {
  int r = row0 + (lane & 15);
  int half = lane >> 4;
  const unsigned short* p = tile + r * ld + koff;
  FragBF f;
#pragma unroll
  for (int i = 0; i < 4; ++i) {
    f.u[i]     = *(const unsigned*)(p + half * 8 + 2 * i);
    f.u[4 + i] = *(const unsigned*)(p + 16 + half * 8 + 2 * i);
  }
  return f.v;
}

// B-fragment (32x16 bf16): lane n = lane&15; VGPR0-3 = K {half*8 .. half*8+7},
// VGPR4-7 = K {16+half*8 .. 16+half*8+7}, 2 K-values per VGPR (matches A)
__device__ __forceinline__ v16bf frag_b(const unsigned short* tile, int ld, int col0,
                                        int koff, int lane) {
  int n = col0 + (lane & 15);
  int half = lane >> 4;
  const unsigned short* p = tile + n * ld + koff;
  FragBF f;
#pragma unroll
  for (int i = 0; i < 4; ++i) {
    f.u[i]     = *(const unsigned*)(p + half * 8 + 2 * i);
    f.u[4 + i] = *(const unsigned*)(p + 16 + half * 8 + 2 * i);
  }
  return f.v;
}

// ---------------------------------------------------------------------------
// Tiled WMMA GEMM:  Out[M,N] = X[M,K] * W[N,K]^T + bias   (optionally ReLU,
// optionally bf16 output).  N multiple of 64; K multiple of 64.
// Block = 128 threads (4 waves), tile 64x64, K-step 64 (2 WMMA k-substeps).
// ---------------------------------------------------------------------------
__global__ __launch_bounds__(128)
void gemm_wmma(const float* __restrict__ X, const float* __restrict__ W,
               const float* __restrict__ bias, void* __restrict__ OutP,
               int M, int N, int K, int relu, int out_bf16) {
  // pitch 68 ushorts = 136 B: keeps 8-byte staging stores aligned, skews banks
  __shared__ unsigned short As[64 * 68];
  __shared__ unsigned short Bs[64 * 68];

  const int m_base = blockIdx.y * 64;
  const int n_base = blockIdx.x * 64;
  const int tid  = threadIdx.x;
  const int lane = tid & 31;
  const int wave = tid >> 5;
  const int wm = (wave >> 1) * 32;   // wave's M offset inside tile
  const int wn = (wave & 1)  * 32;   // wave's N offset inside tile

  v8f acc[2][2];
#pragma unroll
  for (int a = 0; a < 2; ++a)
#pragma unroll
    for (int b = 0; b < 2; ++b)
#pragma unroll
      for (int e = 0; e < 8; ++e) acc[a][b][e] = 0.f;

  for (int k0 = 0; k0 < K; k0 += 64) {
    // prefetch next K-tile (speculative, emits global_prefetch)
    if (k0 + 64 < K) {
      int pm = m_base + (tid & 63);
      if (pm < M) __builtin_prefetch((const void*)(X + (size_t)pm * K + k0 + 64), 0, 3);
      int pn = n_base + (tid & 63);
      if (pn < N) __builtin_prefetch((const void*)(W + (size_t)pn * K + k0 + 64), 0, 3);
    }
    // stage 64x64 tiles (f32 -> bf16) into LDS; b128 loads, row-level guards
    for (int idx = tid; idx < 64 * 16; idx += 128) {
      int r = idx >> 4, c4 = (idx & 15) * 4;
      int gm = m_base + r;
      float4 va = make_float4(0.f, 0.f, 0.f, 0.f);
      if (gm < M) va = *(const float4*)(X + (size_t)gm * K + k0 + c4);
      store_bf4(&As[r * 68 + c4], va);
      int gn = n_base + r;
      float4 vb = make_float4(0.f, 0.f, 0.f, 0.f);
      if (gn < N) vb = *(const float4*)(W + (size_t)gn * K + k0 + c4);
      store_bf4(&Bs[r * 68 + c4], vb);
    }
    __syncthreads();

#pragma unroll
    for (int ks = 0; ks < 64; ks += 32) {
      v16bf a0 = frag_a(As, 68, wm + 0,  ks, lane);
      v16bf a1 = frag_a(As, 68, wm + 16, ks, lane);
      v16bf b0 = frag_b(Bs, 68, wn + 0,  ks, lane);
      v16bf b1 = frag_b(Bs, 68, wn + 16, ks, lane);
      acc[0][0] = __builtin_amdgcn_wmma_f32_16x16x32_bf16(false, a0, false, b0, (short)0, acc[0][0], false, false);
      acc[0][1] = __builtin_amdgcn_wmma_f32_16x16x32_bf16(false, a0, false, b1, (short)0, acc[0][1], false, false);
      acc[1][0] = __builtin_amdgcn_wmma_f32_16x16x32_bf16(false, a1, false, b0, (short)0, acc[1][0], false, false);
      acc[1][1] = __builtin_amdgcn_wmma_f32_16x16x32_bf16(false, a1, false, b1, (short)0, acc[1][1], false, false);
    }
    __syncthreads();
  }

  // C/D layout: VGPR r -> M = r + 8*(lane>=16), N = lane&15  (ISA 7.12.2)
  float* Of = (float*)OutP;
  unsigned short* Oh = (unsigned short*)OutP;
#pragma unroll
  for (int fm = 0; fm < 2; ++fm)
#pragma unroll
    for (int fn = 0; fn < 2; ++fn) {
      int n = n_base + wn + fn * 16 + (lane & 15);
      float bv = bias ? bias[n] : 0.f;
#pragma unroll
      for (int r = 0; r < 8; ++r) {
        int m = m_base + wm + fm * 16 + (lane >> 4) * 8 + r;
        if (m < M) {
          float v = acc[fm][fn][r] + bv;
          if (relu) v = fmaxf(v, 0.f);
          if (out_bf16) Oh[(size_t)m * N + n] = f2bf(v);
          else          Of[(size_t)m * N + n] = v;
        }
      }
    }
}

// ---------------------------------------------------------------------------
// Elementwise add
// ---------------------------------------------------------------------------
__global__ void add_vec(const float* __restrict__ A, const float* __restrict__ B,
                        float* __restrict__ O, int n) {
  int i = blockIdx.x * blockDim.x + threadIdx.x;
  if (i < n) O[i] = A[i] + B[i];
}

// ---------------------------------------------------------------------------
// LayerNorm over C=256:  Out = (Res?) + LN(A (+ B)) * g + beta
// one row per 256-thread block
// ---------------------------------------------------------------------------
__global__ __launch_bounds__(256)
void ln_kernel(const float* __restrict__ A, const float* __restrict__ B,
               const float* __restrict__ Res, const float* __restrict__ g,
               const float* __restrict__ beta, float* __restrict__ Out) {
  int row = blockIdx.x, c = threadIdx.x;
  size_t idx = (size_t)row * 256 + c;
  float v = A[idx] + (B ? B[idx] : 0.f);
  float s = v, s2 = v * v;
#pragma unroll
  for (int o = 16; o > 0; o >>= 1) { s += __shfl_xor(s, o); s2 += __shfl_xor(s2, o); }
  __shared__ float sh1[8], sh2[8];
  int lane = c & 31, wid = c >> 5;
  if (lane == 0) { sh1[wid] = s; sh2[wid] = s2; }
  __syncthreads();
  float ts = 0.f, ts2 = 0.f;
#pragma unroll
  for (int i = 0; i < 8; ++i) { ts += sh1[i]; ts2 += sh2[i]; }
  float mean = ts * (1.f / 256.f);
  float var  = ts2 * (1.f / 256.f) - mean * mean;
  float y = (v - mean) * rsqrtf(var + 1e-5f) * g[c] + beta[c];
  if (Res) y += Res[idx];
  Out[idx] = y;
}

// ---------------------------------------------------------------------------
// Intra attention: 800 blocks (one per (b,nq)), 8 heads, L=16, hd=32
// ---------------------------------------------------------------------------
__global__ __launch_bounds__(128)
void intra_attn(const float* __restrict__ Q, const float* __restrict__ Kb,
                const float* __restrict__ V, float* __restrict__ O) {
  int bq = blockIdx.x;
  __shared__ float qs[16][256], ks[16][256], vs[16][256];
  int tid = threadIdx.x;
  for (int i = tid; i < 16 * 256; i += 128) {
    int l = i >> 8, c = i & 255;
    size_t gidx = ((size_t)bq * 16 + l) * 256 + c;
    qs[l][c] = Q[gidx]; ks[l][c] = Kb[gidx]; vs[l][c] = V[gidx];
  }
  __syncthreads();
  int h = tid >> 4, l = tid & 15;
  const float* qr = &qs[l][h * 32];
  float sc[16], mx = -1e30f;
#pragma unroll
  for (int j = 0; j < 16; ++j) {
    float s = 0.f;
#pragma unroll
    for (int d = 0; d < 32; ++d) s += qr[d] * ks[j][h * 32 + d];
    s *= 0.17677669529663687f;   // 32^-0.5
    sc[j] = s; mx = fmaxf(mx, s);
  }
  float den = 0.f;
#pragma unroll
  for (int j = 0; j < 16; ++j) { sc[j] = __expf(sc[j] - mx); den += sc[j]; }
  float inv = 1.f / den;
#pragma unroll
  for (int d = 0; d < 32; ++d) {
    float o = 0.f;
#pragma unroll
    for (int j = 0; j < 16; ++j) o += sc[j] * vs[j][h * 32 + d];
    O[((size_t)bq * 16 + l) * 256 + h * 32 + d] = o * inv;
  }
}

// ---------------------------------------------------------------------------
// Inter attention: grid (128 = b*16+p, 8 heads), L=100, hd=32, online softmax
// tokens stay in (b, nq, npts) order; permutation folded into indexing.
// ---------------------------------------------------------------------------
__global__ __launch_bounds__(128)
void inter_attn(const float* __restrict__ Q, const float* __restrict__ Kb,
                const float* __restrict__ V, float* __restrict__ O) {
  int bp = blockIdx.x, h = blockIdx.y;
  int b = bp >> 4, p = bp & 15;
  __shared__ float ks[100][32], vs[100][32];
  int tid = threadIdx.x;
  for (int i = tid; i < 100 * 32; i += 128) {
    int j = i >> 5, d = i & 31;
    size_t tok = ((size_t)(b * 100 + j) * 16 + p);
    ks[j][d] = Kb[tok * 256 + h * 32 + d];
    vs[j][d] = V[tok * 256 + h * 32 + d];
  }
  __syncthreads();
  int l = tid;
  if (l < 100) {
    size_t tok = ((size_t)(b * 100 + l) * 16 + p);
    float qr[32], o[32];
#pragma unroll
    for (int d = 0; d < 32; ++d) { qr[d] = Q[tok * 256 + h * 32 + d]; o[d] = 0.f; }
    float mx = -1e30f, den = 0.f;
    for (int j = 0; j < 100; ++j) {
      float s = 0.f;
#pragma unroll
      for (int d = 0; d < 32; ++d) s += qr[d] * ks[j][d];
      s *= 0.17677669529663687f;
      float nmx = fmaxf(mx, s);
      float corr = __expf(mx - nmx);
      float e = __expf(s - nmx);
      den = den * corr + e;
#pragma unroll
      for (int d = 0; d < 32; ++d) o[d] = o[d] * corr + e * vs[j][d];
      mx = nmx;
    }
    float inv = 1.f / den;
#pragma unroll
    for (int d = 0; d < 32; ++d) O[tok * 256 + h * 32 + d] = o[d] * inv;
  }
}

// ---------------------------------------------------------------------------
// Circular conv1d (k=9) over the 16 points + BatchNorm(eval) + ReLU
// one block per (b,nq); thread = output channel
// ---------------------------------------------------------------------------
__global__ __launch_bounds__(256)
void circ_conv(const float* __restrict__ X, const float* __restrict__ CW,
               const float* __restrict__ CB, const float* __restrict__ bg,
               const float* __restrict__ bb, const float* __restrict__ bm,
               const float* __restrict__ bvv, float* __restrict__ O) {
  int bq = blockIdx.x, co = threadIdx.x;
  __shared__ float xs[16][256];
  for (int i = co; i < 4096; i += 256) {
    int l = i >> 8, c = i & 255;
    xs[l][c] = X[((size_t)bq * 16 + l) * 256 + c];
  }
  __syncthreads();
  float acc[16];
  float cb = CB[co];
#pragma unroll
  for (int l = 0; l < 16; ++l) acc[l] = cb;
  const float* wrow = CW + (size_t)co * 2304;   // [C_in=256][9]
  for (int k = 0; k < 9; ++k) {
    for (int ci = 0; ci < 256; ++ci) {
      float w = wrow[ci * 9 + k];
#pragma unroll
      for (int l = 0; l < 16; ++l)
        acc[l] += w * xs[(l + k + 12) & 15][ci];   // (l+k-4) mod 16
    }
  }
  float scale = rsqrtf(bvv[co] + 1e-5f) * bg[co];
  float mmu = bm[co], bof = bb[co];
#pragma unroll
  for (int l = 0; l < 16; ++l) {
    float y = (acc[l] - mmu) * scale + bof;
    O[((size_t)bq * 16 + l) * 256 + co] = fmaxf(y, 0.f);
  }
}

// ---------------------------------------------------------------------------
// MSDA attention-weight softmax over 16 (levels*points), per (token, head)
// ---------------------------------------------------------------------------
__global__ void aw_softmax(float* __restrict__ AW) {
  int i = blockIdx.x * blockDim.x + threadIdx.x;
  if (i >= 12800 * 8) return;
  float* p = AW + (size_t)(i >> 3) * 128 + (i & 7) * 16;
  float mx = -1e30f;
#pragma unroll
  for (int j = 0; j < 16; ++j) mx = fmaxf(mx, p[j]);
  float e[16], s = 0.f;
#pragma unroll
  for (int j = 0; j < 16; ++j) { e[j] = __expf(p[j] - mx); s += e[j]; }
  float inv = 1.f / s;
#pragma unroll
  for (int j = 0; j < 16; ++j) p[j] = e[j] * inv;
}

// ---------------------------------------------------------------------------
// MSDA bilinear sampling + weighted accumulation.
// one block per token (bs*nq*npts = 12800); value stored as bf16 [159576,256]
// ---------------------------------------------------------------------------
__global__ __launch_bounds__(256)
void msda_sample(const unsigned short* __restrict__ Val, const float* __restrict__ Off,
                 const float* __restrict__ AW, const float* __restrict__ Ref,
                 float* __restrict__ O) {
  __shared__ int   rows[128][4];
  __shared__ float wts[128][4];
  const int Wsz[4] = {150, 75, 38, 19};
  const int Hsz[4] = {100, 50, 25, 13};
  const int St[4]  = {0, 15000, 18750, 19700};
  int token = blockIdx.x;
  int tid = threadIdx.x;
  int b = token / 1600;
  int rq = (token % 1600) >> 4;
  if (tid < 128) {
    int h = tid >> 4, s = tid & 15, lvl = s >> 2;
    int W = Wsz[lvl], H = Hsz[lvl];
    float rx = Ref[(((size_t)b * 100 + rq) * 4 + lvl) * 2 + 0];
    float ry = Ref[(((size_t)b * 100 + rq) * 4 + lvl) * 2 + 1];
    float ox = Off[(size_t)token * 256 + h * 32 + s * 2];
    float oy = Off[(size_t)token * 256 + h * 32 + s * 2 + 1];
    float x = (rx + ox / (float)W) * (float)W - 0.5f;
    float y = (ry + oy / (float)H) * (float)H - 0.5f;
    float x0f = floorf(x), y0f = floorf(y);
    float fx = x - x0f, fy = y - y0f;
    int x0 = (int)x0f, y0 = (int)y0f;
    float a = AW[(size_t)token * 128 + h * 16 + s];
    int base = b * 19947 + St[lvl];
#pragma unroll
    for (int c = 0; c < 4; ++c) {
      int dx = c & 1, dy = c >> 1;
      int xi = x0 + dx, yi = y0 + dy;
      bool valid = (xi >= 0) && (xi < W) && (yi >= 0) && (yi < H);
      rows[tid][c] = valid ? (base + yi * W + xi) : -1;
      wts[tid][c] = (dx ? fx : 1.f - fx) * (dy ? fy : 1.f - fy) * a;
    }
  }
  __syncthreads();
  int h = tid >> 5;
  float acc = 0.f;
  for (int s = 0; s < 16; ++s) {
    int sidx = h * 16 + s;
#pragma unroll
    for (int c = 0; c < 4; ++c) {
      int r = rows[sidx][c];
      if (r >= 0) acc += wts[sidx][c] * bf2f(Val[(size_t)r * 256 + tid]);
    }
  }
  O[(size_t)token * 256 + tid] = acc;
}

// ---------------------------------------------------------------------------
// Host launcher
// ---------------------------------------------------------------------------
extern "C" void kernel_launch(void* const* d_in, const int* in_sizes, int n_in,
                              void* d_out, int out_size, void* d_ws, size_t ws_size,
                              hipStream_t stream) {
  (void)in_sizes; (void)n_in; (void)out_size; (void)ws_size;
  const float* tgt        = (const float*)d_in[0];
  const float* qpos       = (const float*)d_in[1];
  const float* refp       = (const float*)d_in[2];
  const float* src        = (const float*)d_in[3];
  const float* wqkv_intra = (const float*)d_in[4];
  const float* bqkv_intra = (const float*)d_in[5];
  const float* wo_intra   = (const float*)d_in[6];
  const float* bo_intra   = (const float*)d_in[7];
  const float* conv_w     = (const float*)d_in[8];
  const float* conv_b     = (const float*)d_in[9];
  const float* bn_g       = (const float*)d_in[10];
  const float* bn_b       = (const float*)d_in[11];
  const float* bn_m       = (const float*)d_in[12];
  const float* bn_v       = (const float*)d_in[13];
  const float* ln_intra_g = (const float*)d_in[14];
  const float* ln_intra_b = (const float*)d_in[15];
  const float* mlp_fuse_w = (const float*)d_in[16];
  const float* mlp_fuse_b = (const float*)d_in[17];
  const float* ln_fuse_g  = (const float*)d_in[18];
  const float* ln_fuse_b  = (const float*)d_in[19];
  const float* wqkv_inter = (const float*)d_in[20];
  const float* bqkv_inter = (const float*)d_in[21];
  const float* wo_inter   = (const float*)d_in[22];
  const float* bo_inter   = (const float*)d_in[23];
  const float* ln_inter_g = (const float*)d_in[24];
  const float* ln_inter_b = (const float*)d_in[25];
  const float* w_off      = (const float*)d_in[26];
  const float* b_off      = (const float*)d_in[27];
  const float* w_attn     = (const float*)d_in[28];
  const float* b_attn     = (const float*)d_in[29];
  const float* w_val      = (const float*)d_in[30];
  const float* b_val      = (const float*)d_in[31];
  const float* w_out      = (const float*)d_in[32];
  const float* b_out      = (const float*)d_in[33];
  const float* ln_cross_g = (const float*)d_in[34];
  const float* ln_cross_b = (const float*)d_in[35];
  const float* w1         = (const float*)d_in[36];
  const float* b1         = (const float*)d_in[37];
  const float* w2         = (const float*)d_in[38];
  const float* b2         = (const float*)d_in[39];
  const float* ln3_g      = (const float*)d_in[40];
  const float* ln3_b      = (const float*)d_in[41];

  float* out = (float*)d_out;
  float* ws  = (float*)d_ws;

  const int TOK = 12800;                 // bs*nq*npts
  const size_t TOKC = (size_t)TOK * 256;
  float* G0 = ws + 0 * TOKC;
  float* G1 = ws + 1 * TOKC;
  float* G2 = ws + 2 * TOKC;
  float* G3 = ws + 3 * TOKC;
  float* G4 = ws + 4 * TOKC;
  float* G5 = ws + 5 * TOKC;
  float* G6 = ws + 6 * TOKC;
  float* G7 = ws + 7 * TOKC;
  unsigned short* Vval = (unsigned short*)(ws + 8 * TOKC);      // [159576,256] bf16
  float* Fffn = ws + 8 * TOKC + (size_t)(159576ull * 256 / 2);  // [12800,1024] f32
  float* AW   = Fffn + (size_t)12800 * 1024;                    // [12800,128]  f32

  auto gemm = [&](const float* X, const float* W, const float* bias, void* Out,
                  int M, int N, int K, int relu, int obf) {
    dim3 grid(N / 64, (M + 63) / 64);
    gemm_wmma<<<grid, 128, 0, stream>>>(X, W, bias, Out, M, N, K, relu, obf);
  };
  const int NBLK = (int)((TOKC + 255) / 256);

  // ---- EFSA intra: qk = tgt + query_pos -----------------------------------
  add_vec<<<NBLK, 256, 0, stream>>>(tgt, qpos, G0, (int)TOKC);
  gemm(G0,  wqkv_intra,           bqkv_intra,       G1, TOK, 256, 256, 0, 0); // q
  gemm(G0,  wqkv_intra + 65536,   bqkv_intra + 256, G2, TOK, 256, 256, 0, 0); // k
  gemm(tgt, wqkv_intra + 131072,  bqkv_intra + 512, G3, TOK, 256, 256, 0, 0); // v
  intra_attn<<<800, 128, 0, stream>>>(G1, G2, G3, G4);
  gemm(G4, wo_intra, bo_intra, G5, TOK, 256, 256, 0, 0);                      // attn proj
  circ_conv<<<800, 256, 0, stream>>>(G0, conv_w, conv_b, bn_g, bn_b, bn_m, bn_v, G6);
  ln_kernel<<<TOK, 256, 0, stream>>>(G5, G6, tgt, ln_intra_g, ln_intra_b, G7); // tgt1
  gemm(G7, mlp_fuse_w, mlp_fuse_b, G4, TOK, 256, 256, 0, 0);
  ln_kernel<<<TOK, 256, 0, stream>>>(G4, nullptr, G7, ln_fuse_g, ln_fuse_b, G0); // tgt2

  // ---- inter attention across instances -----------------------------------
  gemm(G0, wqkv_inter,          bqkv_inter,       G1, TOK, 256, 256, 0, 0);
  gemm(G0, wqkv_inter + 65536,  bqkv_inter + 256, G2, TOK, 256, 256, 0, 0);
  gemm(G0, wqkv_inter + 131072, bqkv_inter + 512, G3, TOK, 256, 256, 0, 0);
  inter_attn<<<dim3(128, 8), 128, 0, stream>>>(G1, G2, G3, G4);
  gemm(G4, wo_inter, bo_inter, G5, TOK, 256, 256, 0, 0);
  ln_kernel<<<TOK, 256, 0, stream>>>(G0, G5, nullptr, ln_inter_g, ln_inter_b, G6); // tgt_inter

  // ---- deformable cross attention -----------------------------------------
  add_vec<<<NBLK, 256, 0, stream>>>(G6, qpos, G1, (int)TOKC);                 // q_cross
  gemm(src, w_val, b_val, Vval, 159576, 256, 256, 0, 1);                      // value (bf16)
  gemm(G1, w_off,  b_off,  G2, TOK, 256, 256, 0, 0);                          // offsets
  gemm(G1, w_attn, b_attn, AW, TOK, 128, 256, 0, 0);                          // attn logits
  aw_softmax<<<(12800 * 8 + 255) / 256, 256, 0, stream>>>(AW);
  msda_sample<<<TOK, 256, 0, stream>>>(Vval, G2, AW, refp, G3);
  gemm(G3, w_out, b_out, G4, TOK, 256, 256, 0, 0);
  ln_kernel<<<TOK, 256, 0, stream>>>(G6, G4, nullptr, ln_cross_g, ln_cross_b, G5); // tgt_cross

  // ---- FFN ----------------------------------------------------------------
  gemm(G5, w1, b1, Fffn, TOK, 1024, 256, 1, 0);                               // relu
  gemm(Fffn, w2, b2, G1, TOK, 256, 1024, 0, 0);
  ln_kernel<<<TOK, 256, 0, stream>>>(G5, G1, nullptr, ln3_g, ln3_b, out);
}
